// GatedBlock_45638322487323
// MI455X (gfx1250) — matrix operand order repaired and checked by reading, not compiled, not run on hardware
//
#include <hip/hip_runtime.h>
#include <math.h>

typedef __attribute__((ext_vector_type(16))) __bf16 v16bf;
typedef __attribute__((ext_vector_type(8)))  __bf16 v8bf;
typedef __attribute__((ext_vector_type(8)))  float  v8f;

#define EPI_BIAS_BF16 0
#define EPI_GELU_BF16 1
#define EPI_GATE_SET  2
#define EPI_GATE_ADD  3

__device__ __forceinline__ v8f wmma_bf16(v16bf a, v16bf b, v8f c) {
  // D = A(16x32 bf16) * B(32x16 bf16) + C(16x16 f32)
  return __builtin_amdgcn_wmma_f32_16x16x32_bf16(false, a, false, b, (short)0, c,
                                                 false, false);
}

__device__ __forceinline__ v16bf cat16(v8bf lo, v8bf hi) {
  return __builtin_shufflevector(lo, hi, 0, 1, 2, 3, 4, 5, 6, 7, 8, 9, 10, 11, 12,
                                 13, 14, 15);
}

__device__ __forceinline__ v8f zf8() {
  v8f z;
#pragma unroll
  for (int i = 0; i < 8; ++i) z[i] = 0.0f;
  return z;
}

__device__ __forceinline__ v8bf zh8() {
  v8bf z;
#pragma unroll
  for (int i = 0; i < 8; ++i) z[i] = (__bf16)0.0f;
  return z;
}

// ---- weight convert + transpose: W[K][N] f32 -> Wt[N][K] bf16 --------------
__global__ __launch_bounds__(256) void convt_kernel(const float* __restrict__ W,
                                                    __bf16* __restrict__ Wt,
                                                    int K, int N) {
  size_t idx = (size_t)blockIdx.x * 256 + threadIdx.x;
  size_t total = (size_t)K * N;
  if (idx >= total) return;
  int k = (int)(idx / N);
  int n = (int)(idx - (size_t)k * N);
  Wt[(size_t)n * K + k] = (__bf16)W[idx];
}

// ---- pool (C -> Q block means) fused with 2-expert softmax gate ------------
__global__ __launch_bounds__(256) void pool_gate_kernel(
    const float* __restrict__ x, const float* __restrict__ wg,
    float* __restrict__ xs, float* __restrict__ gates) {
  __shared__ float r0[256], r1[256];
  const int t = blockIdx.x;  // token 0..2047
  const int n = t >> 6, q = t & 63;
  const float* base = x + ((size_t)n * 512 + (size_t)q * 8) * 2048;
  float s0 = 0.f, s1 = 0.f;
  for (int d = threadIdx.x; d < 2048; d += 256) {
    float sum = 0.f;
#pragma unroll
    for (int j = 0; j < 8; ++j) sum += base[(size_t)j * 2048 + d];
    float m = sum * 0.125f;
    xs[(size_t)t * 2048 + d] = m;
    s0 += m * wg[d * 2 + 0];
    s1 += m * wg[d * 2 + 1];
  }
  r0[threadIdx.x] = s0;
  r1[threadIdx.x] = s1;
  __syncthreads();
  for (int s = 128; s > 0; s >>= 1) {
    if (threadIdx.x < (unsigned)s) {
      r0[threadIdx.x] += r0[threadIdx.x + s];
      r1[threadIdx.x] += r1[threadIdx.x + s];
    }
    __syncthreads();
  }
  if (threadIdx.x == 0) {
    float a = r0[0], b = r1[0];
    float mx = fmaxf(a, b);
    float e0 = __expf(a - mx), e1 = __expf(b - mx);
    float inv = 1.f / (e0 + e1);
    float p0 = e0 * inv, p1 = e1 * inv;
    float den = p0 + p1 + 1e-6f;
    gates[t * 2 + 0] = p0 / den;
    gates[t * 2 + 1] = p1 / den;
  }
}

// ---- generic bf16 WMMA GEMM: out = epilogue(A[MxK] @ Wt[NxK]^T + bias) -----
// Block tile 128x128, BK=64 (2 WMMA K-steps per stage), 256 threads = 8 waves,
// wave tile 32x64 -> 16 WMMAs per barrier pair. Specialized on A dtype + mode.
#define GLDA 72  // 144-byte LDS rows: 16B-aligned, bank-friendly

template <int AF32, int MODE>
__global__ __launch_bounds__(256) void gemm_kernel(
    const void* __restrict__ Aptr, const __bf16* __restrict__ Wt,
    const float* __restrict__ bias, int M, int N, int K,
    const float* __restrict__ gates, int expert, __bf16* __restrict__ outH,
    float* __restrict__ outF) {
  __shared__ __align__(16) __bf16 As[128 * GLDA];
  __shared__ __align__(16) __bf16 Bs[128 * GLDA];
  const int tid = threadIdx.x;
  const int lane = tid & 31, wid = tid >> 5;
  const int l15 = lane & 15, lhalf = lane >> 4;
  const int waveM = wid & 3, waveN = wid >> 2;
  const int m0 = blockIdx.x * 128, n0 = blockIdx.y * 128;

  v8f acc[2][4];
#pragma unroll
  for (int i = 0; i < 2; ++i)
#pragma unroll
    for (int j = 0; j < 4; ++j) acc[i][j] = zf8();

  for (int kt = 0; kt < K; kt += 64) {
    __syncthreads();
    if (AF32) {
      const float* Af = (const float*)Aptr;
#pragma unroll
      for (int i = 0; i < 4; ++i) {
        int ch = tid * 4 + i;              // 1024 chunks of 8 floats
        int row = ch >> 3, col = (ch & 7) << 3;
        int gm = m0 + row;
        v8bf h = zh8();
        if (gm < M) {
          const float4* src = (const float4*)(Af + (size_t)gm * K + kt + col);
          float4 v0 = src[0], v1 = src[1];
          h[0] = (__bf16)v0.x; h[1] = (__bf16)v0.y;
          h[2] = (__bf16)v0.z; h[3] = (__bf16)v0.w;
          h[4] = (__bf16)v1.x; h[5] = (__bf16)v1.y;
          h[6] = (__bf16)v1.z; h[7] = (__bf16)v1.w;
        }
        *(v8bf*)&As[row * GLDA + col] = h;
      }
    } else {
      const __bf16* Ah = (const __bf16*)Aptr;
#pragma unroll
      for (int i = 0; i < 4; ++i) {
        int ch = tid * 4 + i;              // 1024 chunks of 8 bf16
        int row = ch >> 3, col = (ch & 7) << 3;
        int gm = m0 + row;
        v8bf v = zh8();
        if (gm < M) v = *(const v8bf*)(Ah + (size_t)gm * K + kt + col);
        *(v8bf*)&As[row * GLDA + col] = v;
      }
    }
#pragma unroll
    for (int i = 0; i < 4; ++i) {
      int ch = tid * 4 + i;
      int row = ch >> 3, col = (ch & 7) << 3;
      int gn = n0 + row;
      v8bf v = zh8();
      if (gn < N) v = *(const v8bf*)(Wt + (size_t)gn * K + kt + col);
      *(v8bf*)&Bs[row * GLDA + col] = v;
    }
    __syncthreads();

#pragma unroll
    for (int ks = 0; ks < 2; ++ks) {
      const int kk = ks * 32;
      v16bf afr[2], bfr[4];
#pragma unroll
      for (int mt = 0; mt < 2; ++mt) {
        // A frag: elements 0-15 -> K=kk+lhalf*16+0..15
        const v8bf* pa = (const v8bf*)&As[(waveM * 32 + mt * 16 + l15) * GLDA +
                                          kk + lhalf * 16];
        afr[mt] = cat16(pa[0], pa[1]);
      }
#pragma unroll
      for (int nt = 0; nt < 4; ++nt) {
        // B frag: column n=l15, elements 0-15 -> K=kk+lhalf*16+0..15
        const v8bf* pb = (const v8bf*)&Bs[(waveN * 64 + nt * 16 + l15) * GLDA +
                                          kk + lhalf * 16];
        bfr[nt] = cat16(pb[0], pb[1]);
      }
#pragma unroll
      for (int mt = 0; mt < 2; ++mt)
#pragma unroll
        for (int nt = 0; nt < 4; ++nt)
          acc[mt][nt] = wmma_bf16(afr[mt], bfr[nt], acc[mt][nt]);
    }
  }

  // epilogue: C/D layout -> elem r = row (r + lhalf*8), col l15
#pragma unroll
  for (int mt = 0; mt < 2; ++mt) {
#pragma unroll
    for (int nt = 0; nt < 4; ++nt) {
      int gn = n0 + waveN * 64 + nt * 16 + l15;
      float b = (bias != nullptr && gn < N) ? bias[gn] : 0.f;
      int gmb = m0 + waveM * 32 + mt * 16 + lhalf * 8;
#pragma unroll
      for (int r = 0; r < 8; ++r) {
        int gm = gmb + r;
        if (gm >= M || gn >= N) continue;
        float v = acc[mt][nt][r] + b;
        if (MODE == EPI_GELU_BF16) {
          v = 0.5f * v * (1.f + erff(v * 0.70710678118654752f));
          outH[(size_t)gm * N + gn] = (__bf16)v;
        } else if (MODE == EPI_BIAS_BF16) {
          outH[(size_t)gm * N + gn] = (__bf16)v;
        } else {
          float g = gates[gm * 2 + expert];
          float* o = outF + (size_t)gm * N + gn;
          if (MODE == EPI_GATE_SET) *o = g * v;
          else *o = *o + g * v;
        }
      }
    }
  }
}

// ---- cross-attention: one workgroup per (n, h) -----------------------------
#define SLD 520
#define ATTN_Q_BYTES (64 * 136 * 2)
#define ATTN_S_BYTES (64 * SLD * 4)
#define ATTN_P_BYTES (64 * SLD * 2)
#define ATTN_V_BYTES (128 * 72 * 2)
#define ATTN_R_BYTES (512 * 4)
#define ATTN_LDS_BYTES \
  (ATTN_Q_BYTES + ATTN_S_BYTES + ATTN_P_BYTES + ATTN_V_BYTES + ATTN_R_BYTES)

__global__ __launch_bounds__(256) void attn_kernel(const __bf16* __restrict__ qh,
                                                   const __bf16* __restrict__ kh,
                                                   const __bf16* __restrict__ vh,
                                                   __bf16* __restrict__ ctx) {
  extern __shared__ char smem[];
  __bf16* Qs = (__bf16*)smem;                                          // [64][136]
  float* S = (float*)(smem + ATTN_Q_BYTES);                            // [64][SLD]
  __bf16* P = (__bf16*)(smem + ATTN_Q_BYTES + ATTN_S_BYTES);           // [64][SLD]
  __bf16* Vs =
      (__bf16*)(smem + ATTN_Q_BYTES + ATTN_S_BYTES + ATTN_P_BYTES);    // [128][72]
  float* red = (float*)(smem + ATTN_Q_BYTES + ATTN_S_BYTES + ATTN_P_BYTES +
                        ATTN_V_BYTES);

  const int tid = threadIdx.x;
  const int lane = tid & 31, wid = tid >> 5;
  const int l15 = lane & 15, lhalf = lane >> 4;
  const int n = blockIdx.x >> 4, h = blockIdx.x & 15;
  const int qt = wid & 3;    // q-tile 0..3 (16 rows each)
  const int half = wid >> 2; // 0/1: wave pair split

  // stage Q[64][128] slice for this head
#pragma unroll
  for (int i = 0; i < 4; ++i) {
    int ch = tid * 4 + i;
    int row = ch >> 4, col = (ch & 15) << 3;
    *(v8bf*)&Qs[row * 136 + col] =
        *(const v8bf*)(qh + (size_t)row * 2048 + h * 128 + col);
  }
  __syncthreads();

  // phase 1: S = scale * Q @ K^T  (B frags streamed from global K rows)
  const float scale = 0.08838834764831845f;  // 1/sqrt(128)
  for (int ci = 0; ci < 16; ++ci) {
    int ct = half + ci * 2;  // c-tile 0..31
    v8f acc = zf8();
    const __bf16* krow =
        kh + ((size_t)(n * 512 + ct * 16 + l15)) * 2048 + h * 128;
#pragma unroll
    for (int kk = 0; kk < 128; kk += 32) {
      const v8bf* pa = (const v8bf*)&Qs[(qt * 16 + l15) * 136 + kk + lhalf * 16];
      v16bf a = cat16(pa[0], pa[1]);
      const v8bf* pb = (const v8bf*)(krow + kk + lhalf * 16);
      v16bf b = cat16(pb[0], pb[1]);
      acc = wmma_bf16(a, b, acc);
    }
#pragma unroll
    for (int r = 0; r < 8; ++r)
      S[(qt * 16 + lhalf * 8 + r) * SLD + ct * 16 + l15] = acc[r] * scale;
  }
  __syncthreads();

  // phase 2: row softmax over 512 columns (4 threads per row)
  {
    int qrow = tid >> 2, seg = tid & 3;
    float* srow = S + qrow * SLD + seg * 128;
    float mx = -3.0e38f;
    for (int c = 0; c < 128; ++c) mx = fmaxf(mx, srow[c]);
    red[tid] = mx;
    __syncthreads();
    float rmax = fmaxf(fmaxf(red[qrow * 4 + 0], red[qrow * 4 + 1]),
                       fmaxf(red[qrow * 4 + 2], red[qrow * 4 + 3]));
    float sum = 0.f;
    for (int c = 0; c < 128; ++c) sum += __expf(srow[c] - rmax);
    __syncthreads();
    red[tid] = sum;
    __syncthreads();
    float tot = red[qrow * 4 + 0] + red[qrow * 4 + 1] + red[qrow * 4 + 2] +
                red[qrow * 4 + 3];
    float inv = 1.f / tot;
    __bf16* prow = P + qrow * SLD + seg * 128;
    for (int c = 0; c < 128; ++c)
      prow[c] = (__bf16)(__expf(srow[c] - rmax) * inv);
  }
  __syncthreads();

  // phase 3: ctx = P @ V, streaming V^T tiles (64 columns at a time)
  v8f acc3[4];
#pragma unroll
  for (int i = 0; i < 4; ++i) acc3[i] = zf8();
  for (int c0 = 0; c0 < 512; c0 += 64) {
#pragma unroll
    for (int i = 0; i < 4; ++i) {
      int ch = tid * 4 + i;
      int cc = ch >> 4, dc = (ch & 15) << 3;
      v8bf v = *(const v8bf*)(vh + ((size_t)(n * 512 + c0 + cc)) * 2048 +
                              h * 128 + dc);
#pragma unroll
      for (int j = 0; j < 8; ++j) Vs[(dc + j) * 72 + cc] = v[j];
    }
    __syncthreads();
#pragma unroll
    for (int kk = 0; kk < 64; kk += 32) {
      const v8bf* pa =
          (const v8bf*)&P[(qt * 16 + l15) * SLD + c0 + kk + lhalf * 16];
      v16bf a = cat16(pa[0], pa[1]);
#pragma unroll
      for (int dt = 0; dt < 4; ++dt) {
        int dht = half * 4 + dt;
        const v8bf* pb =
            (const v8bf*)&Vs[(dht * 16 + l15) * 72 + kk + lhalf * 16];
        v16bf b = cat16(pb[0], pb[1]);
        acc3[dt] = wmma_bf16(a, b, acc3[dt]);
      }
    }
    __syncthreads();
  }
#pragma unroll
  for (int dt = 0; dt < 4; ++dt) {
    int dh = h * 128 + (half * 4 + dt) * 16 + l15;
#pragma unroll
    for (int r = 0; r < 8; ++r) {
      int qq = qt * 16 + lhalf * 8 + r;
      ctx[((size_t)(n * 64 + qq)) * 2048 + dh] = (__bf16)acc3[dt][r];
    }
  }
}

// ---------------------------------------------------------------------------
extern "C" void kernel_launch(void* const* d_in, const int* in_sizes, int n_in,
                              void* d_out, int out_size, void* d_ws,
                              size_t ws_size, hipStream_t stream) {
  (void)in_sizes; (void)n_in; (void)out_size; (void)ws_size;
  const float* x  = (const float*)d_in[0];
  const float* w1 = (const float*)d_in[1];
  const float* b1 = (const float*)d_in[2];
  const float* w2 = (const float*)d_in[3];
  const float* b2 = (const float*)d_in[4];
  const float* qt = (const float*)d_in[5];
  const float* wq = (const float*)d_in[6];
  const float* bq = (const float*)d_in[7];
  const float* wk = (const float*)d_in[8];
  const float* bk = (const float*)d_in[9];
  const float* wv = (const float*)d_in[10];
  const float* bv = (const float*)d_in[11];
  const float* wo = (const float*)d_in[12];
  const float* bo = (const float*)d_in[13];
  const float* wg = (const float*)d_in[14];
  float* out = (float*)d_out;

  const int D = 2048, NC = 16384, T = 2048;
  char* ws = (char*)d_ws;
  size_t off = 0;
  auto carve = [&](size_t bytes) -> char* {
    char* p = ws + off;
    off += (bytes + 255) & ~(size_t)255;
    return p;
  };
  __bf16* w1T = (__bf16*)carve((size_t)D * D * 2);
  __bf16* w2T = (__bf16*)carve((size_t)D * D * 2);
  __bf16* wqT = (__bf16*)carve((size_t)D * D * 2);
  __bf16* wkT = (__bf16*)carve((size_t)D * D * 2);
  __bf16* wvT = (__bf16*)carve((size_t)D * D * 2);
  __bf16* woT = (__bf16*)carve((size_t)D * D * 2);
  float* xs    = (float*)carve((size_t)T * D * 4);
  float* gates = (float*)carve((size_t)T * 2 * 4);
  __bf16* qh   = (__bf16*)carve((size_t)64 * D * 2);
  __bf16* kh   = (__bf16*)carve((size_t)NC * D * 2);
  __bf16* vh   = (__bf16*)carve((size_t)NC * D * 2);
  __bf16* ctxh = (__bf16*)carve((size_t)T * D * 2);
  __bf16* resh = (__bf16*)carve((size_t)T * D * 2);
  __bf16* hbuf = (__bf16*)carve((size_t)T * D * 2);

  dim3 blk(256);
  const int cgrid = (D * D) / 256;
  convt_kernel<<<cgrid, blk, 0, stream>>>(w1, w1T, D, D);
  convt_kernel<<<cgrid, blk, 0, stream>>>(w2, w2T, D, D);
  convt_kernel<<<cgrid, blk, 0, stream>>>(wq, wqT, D, D);
  convt_kernel<<<cgrid, blk, 0, stream>>>(wk, wkT, D, D);
  convt_kernel<<<cgrid, blk, 0, stream>>>(wv, wvT, D, D);
  convt_kernel<<<cgrid, blk, 0, stream>>>(wo, woT, D, D);

  pool_gate_kernel<<<T, blk, 0, stream>>>(x, wg, xs, gates);

  // q projection (M=64, A fp32)
  gemm_kernel<1, EPI_BIAS_BF16><<<dim3(1, 16), blk, 0, stream>>>(
      qt, wqT, bq, 64, D, D, nullptr, 0, qh, nullptr);
  // K / V projections (M=16384, A fp32)
  gemm_kernel<1, EPI_BIAS_BF16><<<dim3(128, 16), blk, 0, stream>>>(
      x, wkT, bk, NC, D, D, nullptr, 0, kh, nullptr);
  gemm_kernel<1, EPI_BIAS_BF16><<<dim3(128, 16), blk, 0, stream>>>(
      x, wvT, bv, NC, D, D, nullptr, 0, vh, nullptr);
  // cross-attention: 32 batches x 16 heads
  attn_kernel<<<512, blk, ATTN_LDS_BYTES, stream>>>(qh, kh, vh, ctxh);

  // output projection of attention (A bf16)
  gemm_kernel<0, EPI_BIAS_BF16><<<dim3(16, 16), blk, 0, stream>>>(
      ctxh, woT, bo, T, D, D, nullptr, 0, resh, nullptr);
  // expert 0: MLP(pooled), fused gate-combine writes d_out
  gemm_kernel<1, EPI_GELU_BF16><<<dim3(16, 16), blk, 0, stream>>>(
      xs, w1T, b1, T, D, D, nullptr, 0, hbuf, nullptr);
  gemm_kernel<0, EPI_GATE_SET><<<dim3(16, 16), blk, 0, stream>>>(
      hbuf, w2T, b2, T, D, D, gates, 0, nullptr, out);
  // expert 1: MLP(resampled), accumulate into d_out
  gemm_kernel<0, EPI_GELU_BF16><<<dim3(16, 16), blk, 0, stream>>>(
      resh, w1T, b1, T, D, D, nullptr, 0, hbuf, nullptr);
  gemm_kernel<0, EPI_GATE_ADD><<<dim3(16, 16), blk, 0, stream>>>(
      hbuf, w2T, b2, T, D, D, gates, 1, nullptr, out);
}
